// MultiHeadAttention_32865089749139
// MI455X (gfx1250) — matrix operand order repaired and checked by reading, not compiled, not run on hardware
//
#include <hip/hip_runtime.h>
#include <hip/hip_bf16.h>

typedef __attribute__((ext_vector_type(16))) _Float16 v16h;
typedef __attribute__((ext_vector_type(8)))  _Float16 v8h;   // 16B
typedef __attribute__((ext_vector_type(4)))  _Float16 v4h;   // 8B
typedef __attribute__((ext_vector_type(8)))  float    v8f;
typedef __attribute__((__vector_size__(16))) int      i4;    // int4 pointee for async builtin

#if defined(__has_builtin)
#if __has_builtin(__builtin_amdgcn_global_load_async_to_lds_b128)
#define HAVE_ASYNC_LDS 1
#endif
#endif

#define AS1 __attribute__((address_space(1)))
#define AS3 __attribute__((address_space(3)))

// 16-byte global -> LDS copy; async (ASYNCcnt) when available.
__device__ inline void cp16(const _Float16* g, _Float16* l) {
#if defined(HAVE_ASYNC_LDS)
  __builtin_amdgcn_global_load_async_to_lds_b128((AS1 i4*)g, (AS3 i4*)l, 0, 0);
#else
  *(v8h*)l = *(const v8h*)g;
#endif
}

__device__ inline void wait_async_lds() {
#if defined(HAVE_ASYNC_LDS)
#if __has_builtin(__builtin_amdgcn_s_wait_asynccnt)
  __builtin_amdgcn_s_wait_asynccnt(0);
#else
  asm volatile("s_wait_asynccnt 0x0" ::: "memory");
#endif
#endif
}

// D = A(16x32 f16) * B(32x16 f16) + C(16x16 f32)
__device__ inline v8f wmma_f16(v16h a, v16h b, v8f c) {
  return __builtin_amdgcn_wmma_f32_16x16x32_f16(false, a, false, b, (short)0, c, false, false);
}

// ---- Fragment loaders (wave32, CDNA5 WMMA VGPR layouts per ISA 7.12.2) ----
// A 16x32 f16 from row-major [16][lda]: lane L -> row L%16; pairs contiguous.
__device__ inline v16h load_frag_a(const _Float16* p, int lda, int lane) {
  const _Float16* row = p + (lane & 15) * lda;
  int koff = (lane >> 4) << 3;
  v16h f;
#pragma unroll
  for (int v = 0; v < 8; ++v) {
    int k = ((v & 3) << 1) + ((v >> 2) << 4) + koff;
    f[2 * v]     = row[k];
    f[2 * v + 1] = row[k + 1];
  }
  return f;
}

// B 32x16 f16 from n-major tile [16][ldb]: lane L -> col L%16; k = e + 8*(L>=16) (+8 for e>=8).
__device__ inline v16h load_frag_b(const _Float16* p, int ldb, int lane) {
  const _Float16* col = p + (lane & 15) * ldb + ((lane >> 4) << 3);
  v16h f;
#pragma unroll
  for (int v = 0; v < 8; ++v) {
    int k = ((v & 3) << 1) + ((v >> 2) << 4);
    f[2 * v]     = col[k];
    f[2 * v + 1] = col[k + 1];
  }
  return f;
}

// C/D f32 16x16: vgpr r -> (m = r + 8*(L>=16), n = L%16)

// =====================  Kernel 1: fused QKV projection  =====================
#define MT 128
#define NT 128
#define KT 32
#define LDA 36  // padded half stride; 8B-aligned for v4h stores

__global__ __launch_bounds__(256) void qkv_gemm_kernel(
    const float* __restrict__ x, const float* __restrict__ W1,
    const float* __restrict__ b1, _Float16* __restrict__ qws,
    _Float16* __restrict__ kws, _Float16* __restrict__ vtws) {
  __shared__ _Float16 ldsA[MT * LDA];
  __shared__ _Float16 ldsB[NT * LDA];  // transposed: [n][k]
  const int tid = threadIdx.x;
  const int lane = tid & 31;
  const int w = tid >> 5;
  const int m0 = (w >> 2) * 64;
  const int n0 = (w & 3) * 32;
  const int gm = blockIdx.x * MT;
  const int gn = blockIdx.y * NT;

  // register staging for software pipeline (A: 128x32, B: 32x128, float4 each)
  float4 ra[4], rb[4];

#define LOAD_TILES(k0)                                                      \
  {                                                                         \
    _Pragma("unroll") for (int it = 0; it < 4; ++it) {                      \
      int id = tid + it * 256;                                              \
      int r_ = id >> 3, c4 = (id & 7) << 2;                                 \
      ra[it] = *(const float4*)&x[(size_t)(gm + r_) * 1024 + (k0) + c4];    \
      int kk = id >> 5, n_ = (id & 31) << 2;                                \
      rb[it] = *(const float4*)&W1[(size_t)((k0) + kk) * 3072 + gn + n_];   \
    }                                                                       \
  }

#define STORE_TILES()                                                       \
  {                                                                         \
    _Pragma("unroll") for (int it = 0; it < 4; ++it) {                      \
      int id = tid + it * 256;                                              \
      int r_ = id >> 3, c4 = (id & 7) << 2;                                 \
      v4h a_;                                                               \
      a_[0] = (_Float16)ra[it].x; a_[1] = (_Float16)ra[it].y;               \
      a_[2] = (_Float16)ra[it].z; a_[3] = (_Float16)ra[it].w;               \
      *(v4h*)&ldsA[r_ * LDA + c4] = a_;                                     \
      int kk = id >> 5, n_ = (id & 31) << 2;                                \
      ldsB[(n_ + 0) * LDA + kk] = (_Float16)rb[it].x;                       \
      ldsB[(n_ + 1) * LDA + kk] = (_Float16)rb[it].y;                       \
      ldsB[(n_ + 2) * LDA + kk] = (_Float16)rb[it].z;                       \
      ldsB[(n_ + 3) * LDA + kk] = (_Float16)rb[it].w;                       \
    }                                                                       \
  }

  v8f acc[4][2];
#pragma unroll
  for (int i = 0; i < 4; ++i)
#pragma unroll
    for (int j = 0; j < 2; ++j)
#pragma unroll
      for (int r = 0; r < 8; ++r) acc[i][j][r] = 0.0f;

  LOAD_TILES(0);
  for (int k0 = 0; k0 < 1024; k0 += KT) {
    STORE_TILES();
    __syncthreads();
    if (k0 + KT < 1024) LOAD_TILES(k0 + KT);  // prefetch next, hidden under WMMA
    v16h af[4], bf[2];
#pragma unroll
    for (int i = 0; i < 4; ++i)
      af[i] = load_frag_a(&ldsA[(m0 + 16 * i) * LDA], LDA, lane);
#pragma unroll
    for (int j = 0; j < 2; ++j)
      bf[j] = load_frag_b(&ldsB[(n0 + 16 * j) * LDA], LDA, lane);
#pragma unroll
    for (int i = 0; i < 4; ++i)
#pragma unroll
      for (int j = 0; j < 2; ++j) acc[i][j] = wmma_f16(af[i], bf[j], acc[i][j]);
    __syncthreads();
  }

  // Epilogue: bias, fold 1/sqrt(1024) into q, scatter head-major f16 buffers
  const int nl = lane & 15;
  const int mb = (lane >> 4) << 3;
#pragma unroll
  for (int j = 0; j < 2; ++j) {
    int n_g = gn + n0 + 16 * j + nl;  // 0..3071
    float bias = b1[n_g];
    int sel = n_g >> 10;
    int rem = n_g & 1023;
    int h = rem >> 6, d = rem & 63;
#pragma unroll
    for (int i = 0; i < 4; ++i) {
#pragma unroll
      for (int r = 0; r < 8; ++r) {
        int m_g = gm + m0 + 16 * i + mb + r;
        int b = m_g >> 11, t = m_g & 2047;
        int bh = b * 16 + h;
        float v = acc[i][j][r] + bias;
        if (sel == 0)
          qws[((size_t)bh * 2048 + t) * 64 + d] = (_Float16)(v * 0.03125f);
        else if (sel == 1)
          kws[((size_t)bh * 2048 + t) * 64 + d] = (_Float16)v;
        else
          vtws[((size_t)bh * 64 + d) * 2048 + t] = (_Float16)v;
      }
    }
  }
#undef LOAD_TILES
#undef STORE_TILES
}

// =====================  Kernel 2: flash attention  =====================
// Double-buffered K/V LDS tiles filled with async global->LDS B128 copies.
#define LDK 72  // halfs; 144B rows (16B aligned chunks, distinct banks mod 64)
#define LDV 40  // halfs; 80B rows

__global__ __launch_bounds__(256) void attn_kernel(
    const _Float16* __restrict__ q, const _Float16* __restrict__ k,
    const _Float16* __restrict__ vt, float* __restrict__ sdp) {
  __shared__ _Float16 ldsK[2 * 32 * LDK];
  __shared__ _Float16 ldsV[2 * 64 * LDV];
  __shared__ _Float16 ldsP[8 * 16 * 36];

  const int tid = threadIdx.x;
  const int lane = tid & 31;
  const int w = tid >> 5;
  const int bh = blockIdx.x;
  const int qbase = blockIdx.y * 128;

  const _Float16* qp = q + ((size_t)bh * 2048 + qbase + w * 16) * 64;
  const _Float16* kp = k + (size_t)bh * 2048 * 64;
  const _Float16* vp = vt + (size_t)bh * 64 * 2048;

  v16h qf0 = load_frag_a(qp, 64, lane);
  v16h qf1 = load_frag_a(qp + 32, 64, lane);

  v8f acc[4];
  float m_i[8], l_i[8];
#pragma unroll
  for (int t = 0; t < 4; ++t)
#pragma unroll
    for (int r = 0; r < 8; ++r) acc[t][r] = 0.0f;
#pragma unroll
  for (int r = 0; r < 8; ++r) { m_i[r] = -3.0e38f; l_i[r] = 0.0f; }

  const int nl = lane & 15;
  const int mb = (lane >> 4) << 3;
  _Float16* pw = &ldsP[w * 16 * 36];

  // per-thread copy assignments (one 16B chunk each for K and V per tile)
  const int ktk = tid >> 3, kc = (tid & 7) << 3;   // K: 32 rows x 8 chunks
  const int vd = tid >> 2, vc = (tid & 3) << 3;    // V: 64 rows x 4 chunks

#define STAGE(kb, buf)                                                      \
  {                                                                         \
    _Float16* K_ = ldsK + (buf) * (32 * LDK);                               \
    _Float16* V_ = ldsV + (buf) * (64 * LDV);                               \
    cp16(&kp[(size_t)((kb) + ktk) * 64 + kc], &K_[ktk * LDK + kc]);         \
    cp16(&vp[(size_t)vd * 2048 + (kb) + vc], &V_[vd * LDV + vc]);           \
  }

  STAGE(0, 0);
  for (int kb = 0; kb < 2048; kb += 32) {
    const int buf = (kb >> 5) & 1;
    wait_async_lds();
    __syncthreads();
    if (kb + 32 < 2048) STAGE(kb + 32, buf ^ 1);  // prefetch next block
    const _Float16* K = ldsK + buf * (32 * LDK);
    const _Float16* V = ldsV + buf * (64 * LDV);

    v8f s0, s1;
#pragma unroll
    for (int r = 0; r < 8; ++r) { s0[r] = 0.0f; s1[r] = 0.0f; }
    s0 = wmma_f16(qf0, load_frag_b(&K[0], LDK, lane), s0);
    s0 = wmma_f16(qf1, load_frag_b(&K[32], LDK, lane), s0);
    s1 = wmma_f16(qf0, load_frag_b(&K[16 * LDK], LDK, lane), s1);
    s1 = wmma_f16(qf1, load_frag_b(&K[16 * LDK + 32], LDK, lane), s1);

    float corr[8];
#pragma unroll
    for (int r = 0; r < 8; ++r) {
      float mx = fmaxf(s0[r], s1[r]);
      mx = fmaxf(mx, __shfl_xor(mx, 1, 32));
      mx = fmaxf(mx, __shfl_xor(mx, 2, 32));
      mx = fmaxf(mx, __shfl_xor(mx, 4, 32));
      mx = fmaxf(mx, __shfl_xor(mx, 8, 32));
      float mnew = fmaxf(m_i[r], mx);
      float c = __expf(m_i[r] - mnew);
      float a0 = __expf(s0[r] - mnew);
      float a1 = __expf(s1[r] - mnew);
      float rs = a0 + a1;
      rs += __shfl_xor(rs, 1, 32);
      rs += __shfl_xor(rs, 2, 32);
      rs += __shfl_xor(rs, 4, 32);
      rs += __shfl_xor(rs, 8, 32);
      l_i[r] = l_i[r] * c + rs;
      m_i[r] = mnew;
      corr[r] = c;
      pw[(mb + r) * 36 + nl] = (_Float16)a0;
      pw[(mb + r) * 36 + 16 + nl] = (_Float16)a1;
    }
    v16h pf = load_frag_a(pw, 36, lane);
#pragma unroll
    for (int t = 0; t < 4; ++t) {
#pragma unroll
      for (int r = 0; r < 8; ++r) acc[t][r] *= corr[r];
      acc[t] = wmma_f16(pf, load_frag_b(&V[16 * t * LDV], LDV, lane), acc[t]);
    }
  }
#undef STAGE

  const int b = bh >> 4, h = bh & 15;
#pragma unroll
  for (int t = 0; t < 4; ++t) {
#pragma unroll
    for (int r = 0; r < 8; ++r) {
      int tg = qbase + w * 16 + mb + r;
      int c = h * 64 + t * 16 + nl;
      sdp[((size_t)(b * 2048 + tg)) * 1024 + c] = acc[t][r] / l_i[r];
    }
  }
}

// =====================  Kernel 3: residual + LayerNorm  =====================
__global__ __launch_bounds__(256) void resid_ln_kernel(
    const float* __restrict__ x, const float* __restrict__ sdp,
    const float* __restrict__ gamma, const float* __restrict__ beta,
    float* __restrict__ out) {
  __shared__ float ss[8], ssq[8];
  const int row = blockIdx.x;
  const int tid = threadIdx.x;
  const float* xr = x + (size_t)row * 1024;
  const float* sr = sdp + (size_t)row * 1024;
  float y[4], s = 0.0f, sq = 0.0f;
#pragma unroll
  for (int i = 0; i < 4; ++i) {
    int c = tid + i * 256;
    float v = xr[c] + sr[c];
    y[i] = v; s += v; sq += v * v;
  }
#pragma unroll
  for (int mask = 1; mask < 32; mask <<= 1) {
    s += __shfl_xor(s, mask, 32);
    sq += __shfl_xor(sq, mask, 32);
  }
  if ((tid & 31) == 0) { ss[tid >> 5] = s; ssq[tid >> 5] = sq; }
  __syncthreads();
  s = 0.0f; sq = 0.0f;
#pragma unroll
  for (int j = 0; j < 8; ++j) { s += ss[j]; sq += ssq[j]; }
  float mean = s * (1.0f / 1024.0f);
  float var = sq * (1.0f / 1024.0f) - mean * mean;
  float rstd = rsqrtf(var + 1e-6f);
#pragma unroll
  for (int i = 0; i < 4; ++i) {
    int c = tid + i * 256;
    out[(size_t)row * 1024 + c] = (y[i] - mean) * rstd * gamma[c] + beta[c];
  }
}

extern "C" void kernel_launch(void* const* d_in, const int* in_sizes, int n_in,
                              void* d_out, int out_size, void* d_ws, size_t ws_size,
                              hipStream_t stream) {
  const float* x     = (const float*)d_in[0];  // [2,2048,1024]
  const float* W1    = (const float*)d_in[1];  // [1024,3072]
  const float* b1    = (const float*)d_in[2];  // [3072]
  const float* gamma = (const float*)d_in[3];  // [1024]
  const float* beta  = (const float*)d_in[4];  // [1024]
  float* out = (float*)d_out;

  char* ws = (char*)d_ws;
  const size_t QKV_HALFS = (size_t)2 * 16 * 2048 * 64;
  _Float16* qws  = (_Float16*)(ws);
  _Float16* kws  = (_Float16*)(ws + QKV_HALFS * 2);
  _Float16* vtws = (_Float16*)(ws + QKV_HALFS * 4);
  float*    sdp  = (float*)   (ws + QKV_HALFS * 6);

  qkv_gemm_kernel<<<dim3(4096 / MT, 3072 / NT), 256, 0, stream>>>(
      x, W1, b1, qws, kws, vtws);
  attn_kernel<<<dim3(32, 16), 256, 0, stream>>>(qws, kws, vtws, sdp);
  resid_ln_kernel<<<4096, 256, 0, stream>>>(x, sdp, gamma, beta, out);
}
